// LocalAttentionBlock_38491496907023
// MI455X (gfx1250) — matrix operand order repaired and checked by reading, not compiled, not run on hardware
//
#include <hip/hip_runtime.h>
#include <hip/hip_bf16.h>
#include <math.h>

typedef __attribute__((ext_vector_type(16))) __bf16 v16bf;
typedef __attribute__((ext_vector_type(8)))  float  v8f;

#define LN_10000 9.21034037f

// ---------------------------------------------------------------------------
// WMMA helper: D = A(16x32 bf16) * B(32x16 bf16) + C(16x16 f32)
// ---------------------------------------------------------------------------
__device__ __forceinline__ v8f wmma_bf16(v16bf a, v16bf b, v8f c) {
  return __builtin_amdgcn_wmma_f32_16x16x32_bf16(false, a, false, b,
                                                 (short)0, c, false, false);
}

// A fragment (16x32, MxK), row-major source with leading dim lda (elements).
// lanes 0-15 hold M=0..15 with K=0..15; lanes 16-31 hold K=16..31.
// One contiguous 16-element run per lane -> 2x ds_load_b128.
__device__ __forceinline__ v16bf load_a_frag(const __bf16* a, int lda) {
  const int lane = threadIdx.x & 31;
  const int m = lane & 15;
  const int gl = lane >> 4;
  v16bf r;
#pragma unroll
  for (int v = 0; v < 8; ++v) {
    const int k = 16 * gl + 2 * v;
    r[2 * v]     = a[m * lda + k];
    r[2 * v + 1] = a[m * lda + k + 1];
  }
  return r;
}

// B fragment where source is stored N-major: element(k,n) = p[n*ld + k].
// One contiguous 16-element run per lane -> 2x ds_load_b128.
__device__ __forceinline__ v16bf load_bT_frag(const __bf16* p, int ld) {
  const int lane = threadIdx.x & 31;
  const int n = lane & 15;
  const int kg = (lane >> 4) * 16;
  v16bf r;
#pragma unroll
  for (int v = 0; v < 8; ++v) {
    r[2 * v]     = p[n * ld + kg + 2 * v];
    r[2 * v + 1] = p[n * ld + kg + 2 * v + 1];
  }
  return r;
}

// ---------------------------------------------------------------------------
// Elementwise conversion kernels
// ---------------------------------------------------------------------------
__global__ void cvt_f32_bf16(const float* __restrict__ in,
                             __bf16* __restrict__ out, long n) {
  long i = (long)blockIdx.x * blockDim.x + threadIdx.x;
  if (i < n) out[i] = (__bf16)in[i];
}

__global__ void pack_kv_bf16(const float* __restrict__ wk,
                             const float* __restrict__ wv,
                             __bf16* __restrict__ wkv, int Wd, int Hd) {
  long i = (long)blockIdx.x * blockDim.x + threadIdx.x;
  long total = (long)Wd * 2 * Hd;
  if (i >= total) return;
  int c = (int)(i % (2 * Hd));
  long w = i / (2 * Hd);
  float v = (c < Hd) ? wk[w * Hd + c] : wv[w * Hd + (c - Hd)];
  wkv[i] = (__bf16)v;
}

__global__ void seg_scan(const int* __restrict__ segpos,
                         int* __restrict__ segstart, int T) {
  if (threadIdx.x != 0) return;
  int b = blockIdx.x;
  int cur = 0;
  for (int t = 0; t < T; ++t) {
    if (segpos[b * T + t] == 0) cur = t;
    segstart[b * T + t] = cur;
  }
}

// ---------------------------------------------------------------------------
// GEMM: C(MxN f32) = A(MxK bf16) * B(KxN bf16) [+ bias]
// 256 thr (8 waves), tile 128x128, K-step 32, ping-pong LDS + reg prefetch.
// ---------------------------------------------------------------------------
#define G_BM 128
#define G_BN 128
#define G_BK 32
#define G_LDA 48
#define G_LDT 40

__global__ __launch_bounds__(256) void gemm_bf16_f32(
    const __bf16* __restrict__ A, const __bf16* __restrict__ B,
    const float* __restrict__ bias, float* __restrict__ C,
    int M, int N, int K) {
  __shared__ __bf16 As[2][G_BM * G_LDA];
  __shared__ __bf16 BsT[2][G_BN * G_LDT];   // [n][k]

  const int tid = threadIdx.x;
  const int lane = tid & 31;
  const int wave = tid >> 5;
  const int m0 = blockIdx.y * G_BM;
  const int n0 = blockIdx.x * G_BN;
  const int mw = wave * 16;
  const int nl = lane & 15;
  const int mh = (lane >> 4) * 8;

  // per-thread cooperative-load coordinates (2 uint4 per matrix per tile)
  const int arow0 = tid >> 2, arow1 = arow0 + 64, ac8 = tid & 3;
  const int brow0 = tid >> 4, brow1 = brow0 + 16, bc8 = tid & 15;

  v8f acc[8];
#pragma unroll
  for (int i = 0; i < 8; ++i)
#pragma unroll
    for (int j = 0; j < 8; ++j) acc[i][j] = 0.f;

  // prefetch tile 0 into registers
  uint4 ra0 = *(const uint4*)(A + (long)(m0 + arow0) * K + ac8 * 8);
  uint4 ra1 = *(const uint4*)(A + (long)(m0 + arow1) * K + ac8 * 8);
  uint4 rb0 = *(const uint4*)(B + (long)brow0 * N + n0 + bc8 * 8);
  uint4 rb1 = *(const uint4*)(B + (long)brow1 * N + n0 + bc8 * 8);

  int p = 0;
  for (int kt = 0; kt < K; kt += G_BK) {
    // commit prefetched tile to LDS (B transposed via u16 scatter)
    *(uint4*)(&As[p][arow0 * G_LDA + ac8 * 8]) = ra0;
    *(uint4*)(&As[p][arow1 * G_LDA + ac8 * 8]) = ra1;
    {
      __bf16 t0[8], t1[8];
      *(uint4*)t0 = rb0;
      *(uint4*)t1 = rb1;
#pragma unroll
      for (int j = 0; j < 8; ++j) {
        BsT[p][(bc8 * 8 + j) * G_LDT + brow0] = t0[j];
        BsT[p][(bc8 * 8 + j) * G_LDT + brow1] = t1[j];
      }
    }
    __syncthreads();

    // issue next tile's global loads before compute (latency overlap)
    const int ktn = kt + G_BK;
    if (ktn < K) {
      ra0 = *(const uint4*)(A + (long)(m0 + arow0) * K + ktn + ac8 * 8);
      ra1 = *(const uint4*)(A + (long)(m0 + arow1) * K + ktn + ac8 * 8);
      rb0 = *(const uint4*)(B + (long)(ktn + brow0) * N + n0 + bc8 * 8);
      rb1 = *(const uint4*)(B + (long)(ktn + brow1) * N + n0 + bc8 * 8);
    }

    v16bf af = load_a_frag(&As[p][mw * G_LDA], G_LDA);
#pragma unroll
    for (int nt = 0; nt < 8; ++nt) {
      v16bf bf_ = load_bT_frag(&BsT[p][(nt * 16) * G_LDT], G_LDT);
      acc[nt] = wmma_bf16(af, bf_, acc[nt]);
    }
    p ^= 1;
  }

  // epilogue: hoisted bias, pointer-incremented stores
  float bv[8];
#pragma unroll
  for (int nt = 0; nt < 8; ++nt) bv[nt] = 0.f;
  if (bias) {
#pragma unroll
    for (int nt = 0; nt < 8; ++nt) bv[nt] = bias[n0 + nt * 16 + nl];
  }
  float* cbase = C + (long)(m0 + mw + mh) * N + n0 + nl;
#pragma unroll
  for (int r = 0; r < 8; ++r) {
    float* cr = cbase + (long)r * N;
#pragma unroll
    for (int nt = 0; nt < 8; ++nt) cr[nt * 16] = acc[nt][r] + bv[nt];
  }
}

// ---------------------------------------------------------------------------
// RoPE for q: f32 [BT,N,H] -> bf16 [BT,N,H], rope on first H/2, scale folded.
// ---------------------------------------------------------------------------
__global__ void rope_q_kernel(const float* __restrict__ qf,
                              const int* __restrict__ segpos,
                              __bf16* __restrict__ qb,
                              long BT, int Nh, int Hd, float scale) {
  long i = (long)blockIdx.x * blockDim.x + threadIdx.x;
  const int half = Hd / 2, qr = Hd / 4;
  long total = BT * Nh * half;
  if (i >= total) return;
  int j = (int)(i % half);
  long btn = i / half;
  int n = (int)(btn % Nh);
  long bt = btn / Nh;
  const float* src = qf + (bt * Nh + n) * Hd;
  __bf16* dst = qb + (bt * Nh + n) * Hd;
  if (j < qr) {
    float pos = (float)segpos[bt];
    float inv = __expf(-LN_10000 * (float)(2 * j) / (float)half);
    float ang = pos * inv;
    float s = __sinf(ang), c = __cosf(ang);
    float x1 = src[j], x2 = src[j + qr];
    dst[j]      = (__bf16)((x1 * c - x2 * s) * scale);
    dst[j + qr] = (__bf16)((x2 * c + x1 * s) * scale);
  } else {
    int h = j + qr;
    dst[h]      = (__bf16)(src[h] * scale);
    dst[h + qr] = (__bf16)(src[h + qr] * scale);
  }
}

__global__ void rope_kv_kernel(const float* __restrict__ kvf,
                               const int* __restrict__ segpos,
                               __bf16* __restrict__ kb,
                               __bf16* __restrict__ vb,
                               long BT, int Hd) {
  long i = (long)blockIdx.x * blockDim.x + threadIdx.x;
  const int half = Hd / 2, qr = Hd / 4;
  long total = BT * half;
  if (i >= total) return;
  int j = (int)(i % half);
  long bt = i / half;
  const float* src = kvf + bt * 2 * Hd;
  __bf16* kd = kb + bt * Hd;
  __bf16* vd = vb + bt * Hd;
  if (j < qr) {
    float pos = (float)segpos[bt];
    float inv = __expf(-LN_10000 * (float)(2 * j) / (float)half);
    float ang = pos * inv;
    float s = __sinf(ang), c = __cosf(ang);
    float x1 = src[j], x2 = src[j + qr];
    kd[j]      = (__bf16)(x1 * c - x2 * s);
    kd[j + qr] = (__bf16)(x2 * c + x1 * s);
  } else {
    int h = j + qr;
    kd[h]      = (__bf16)src[h];
    kd[h + qr] = (__bf16)src[h + qr];
  }
  vd[2 * j]     = (__bf16)src[Hd + 2 * j];
  vd[2 * j + 1] = (__bf16)src[Hd + 2 * j + 1];
}

// ---------------------------------------------------------------------------
// Windowed-causal flash attention (MQA). 256 thr = 8 waves; wave = one head
// x 16 queries; waves share ping-pong K/V 32-key LDS chunks + reg prefetch.
// grid: (T/16, N/8, B)
// ---------------------------------------------------------------------------
#define A_HD 128
#define A_LDV 40

__global__ __launch_bounds__(256) void attn_kernel(
    const __bf16* __restrict__ qb, const __bf16* __restrict__ kb,
    const __bf16* __restrict__ vb, const int* __restrict__ segstart,
    const int* __restrict__ winp, __bf16* __restrict__ enc,
    int T, int Nh) {
  __shared__ __bf16 Ks[2][32 * A_HD];       // [key][hdim]
  __shared__ __bf16 VsT[2][A_HD * A_LDV];   // [hdim][key]
  __shared__ __bf16 Ps[8][16 * 32];

  const int tid = threadIdx.x;
  const int lane = tid & 31;
  const int wave = tid >> 5;
  const int nl = lane & 15;
  const int mh = (lane >> 4) * 8;
  const int q0 = blockIdx.x * 16;
  const int head = blockIdx.y * 8 + wave;
  const int b = blockIdx.z;
  const int window = winp[0];
  const int W = Nh * A_HD;

  const int krow0 = tid >> 4, krow1 = krow0 + 16, c8 = tid & 15;

  // q fragments (done once per wave)
  const __bf16* qptr = qb + ((long)(b * T + q0) * Nh + head) * A_HD;
  v16bf qf[4];
#pragma unroll
  for (int hk = 0; hk < 4; ++hk) qf[hk] = load_a_frag(qptr + hk * 32, W);

  int segs[8];
#pragma unroll
  for (int r = 0; r < 8; ++r) segs[r] = segstart[b * T + q0 + mh + r];

  float m_[8], l_[8];
  v8f O[8];
#pragma unroll
  for (int r = 0; r < 8; ++r) { m_[r] = -3.0e38f; l_[r] = 0.f; }
#pragma unroll
  for (int i = 0; i < 8; ++i)
#pragma unroll
    for (int j = 0; j < 8; ++j) O[i][j] = 0.f;

  int kstart = q0 - window; if (kstart < 0) kstart = 0;
  kstart &= ~31;
  const int nch = (q0 + 15 - kstart) / 32 + 1;

  const uint4 zz = {0u, 0u, 0u, 0u};
  const __bf16* kbb = kb + (long)(b * T) * A_HD;
  const __bf16* vbb = vb + (long)(b * T) * A_HD;

  // prefetch chunk 0
  uint4 vk0 = zz, vk1 = zz, vv0 = zz, vv1 = zz;
  {
    int key0 = kstart + krow0, key1 = kstart + krow1;
    if (key0 < T) { vk0 = *(const uint4*)(kbb + (long)key0 * A_HD + c8 * 8);
                    vv0 = *(const uint4*)(vbb + (long)key0 * A_HD + c8 * 8); }
    if (key1 < T) { vk1 = *(const uint4*)(kbb + (long)key1 * A_HD + c8 * 8);
                    vv1 = *(const uint4*)(vbb + (long)key1 * A_HD + c8 * 8); }
  }

  int p = 0;
  for (int c = 0; c < nch; ++c) {
    const int kc = kstart + c * 32;

    // commit prefetched chunk to LDS (V transposed)
    *(uint4*)(&Ks[p][krow0 * A_HD + c8 * 8]) = vk0;
    *(uint4*)(&Ks[p][krow1 * A_HD + c8 * 8]) = vk1;
    {
      __bf16 t0[8], t1[8];
      *(uint4*)t0 = vv0;
      *(uint4*)t1 = vv1;
#pragma unroll
      for (int j = 0; j < 8; ++j) {
        VsT[p][(c8 * 8 + j) * A_LDV + krow0] = t0[j];
        VsT[p][(c8 * 8 + j) * A_LDV + krow1] = t1[j];
      }
    }
    __syncthreads();

    // prefetch next chunk
    if (c + 1 < nch) {
      int key0 = kc + 32 + krow0, key1 = kc + 32 + krow1;
      vk0 = vk1 = vv0 = vv1 = zz;
      if (key0 < T) { vk0 = *(const uint4*)(kbb + (long)key0 * A_HD + c8 * 8);
                      vv0 = *(const uint4*)(vbb + (long)key0 * A_HD + c8 * 8); }
      if (key1 < T) { vk1 = *(const uint4*)(kbb + (long)key1 * A_HD + c8 * 8);
                      vv1 = *(const uint4*)(vbb + (long)key1 * A_HD + c8 * 8); }
    }

    // S = q . k^T for 32 keys (two 16x16 tiles)
    v8f S[2];
#pragma unroll
    for (int t2 = 0; t2 < 2; ++t2) {
      v8f s;
#pragma unroll
      for (int j = 0; j < 8; ++j) s[j] = 0.f;
#pragma unroll
      for (int hk = 0; hk < 4; ++hk) {
        v16bf bf_ = load_bT_frag(&Ks[p][(t2 * 16) * A_HD + hk * 32], A_HD);
        s = wmma_bf16(qf[hk], bf_, s);
      }
#pragma unroll
      for (int r = 0; r < 8; ++r) {
        int qi = q0 + mh + r;
        int ki = kc + t2 * 16 + nl;
        bool valid = (ki <= qi) && (ki >= qi - window) && (ki >= segs[r]);
        if (!valid) s[r] = -3.0e38f;
      }
      S[t2] = s;
    }

    // online softmax (row reductions across 16-lane half)
    float alpha[8];
#pragma unroll
    for (int r = 0; r < 8; ++r) {
      float v = fmaxf(S[0][r], S[1][r]);
      v = fmaxf(v, __shfl_xor(v, 1));
      v = fmaxf(v, __shfl_xor(v, 2));
      v = fmaxf(v, __shfl_xor(v, 4));
      v = fmaxf(v, __shfl_xor(v, 8));
      float nm = fmaxf(m_[r], v);
      alpha[r] = __expf(m_[r] - nm);
      m_[r] = nm;
    }
#pragma unroll
    for (int ht = 0; ht < 8; ++ht)
#pragma unroll
      for (int r = 0; r < 8; ++r) O[ht][r] *= alpha[r];

#pragma unroll
    for (int t2 = 0; t2 < 2; ++t2)
#pragma unroll
      for (int r = 0; r < 8; ++r) {
        float s = S[t2][r];
        float pexp = (s < -1.0e37f) ? 0.f : __expf(s - m_[r]);
        S[t2][r] = pexp;
        Ps[wave][(mh + r) * 32 + t2 * 16 + nl] = (__bf16)pexp;
      }
#pragma unroll
    for (int r = 0; r < 8; ++r) {
      float v = S[0][r] + S[1][r];
      v += __shfl_xor(v, 1);
      v += __shfl_xor(v, 2);
      v += __shfl_xor(v, 4);
      v += __shfl_xor(v, 8);
      l_[r] = l_[r] * alpha[r] + v;
    }

    // O += P (16x32) @ V (32x128)
    v16bf pf = load_a_frag(&Ps[wave][0], 32);
#pragma unroll
    for (int ht = 0; ht < 8; ++ht) {
      v16bf bf_ = load_bT_frag(&VsT[p][(ht * 16) * A_LDV], A_LDV);
      O[ht] = wmma_bf16(pf, bf_, O[ht]);
    }
    p ^= 1;
  }

  // write encoded [b, t, n*H + h] as bf16
  __bf16* ebase = enc + (long)(b * T + q0 + mh) * W + head * A_HD + nl;
#pragma unroll
  for (int r = 0; r < 8; ++r) {
    __bf16* er = ebase + (long)r * W;
    float il = 1.0f / l_[r];
#pragma unroll
    for (int ht = 0; ht < 8; ++ht) er[ht * 16] = (__bf16)(O[ht][r] * il);
  }
}

// ---------------------------------------------------------------------------
// Host-side launch
// ---------------------------------------------------------------------------
extern "C" void kernel_launch(void* const* d_in, const int* in_sizes, int n_in,
                              void* d_out, int out_size, void* d_ws, size_t ws_size,
                              hipStream_t stream) {
  const float* x     = (const float*)d_in[0];
  const float* w_q   = (const float*)d_in[1];
  const float* w_k   = (const float*)d_in[2];
  const float* w_v   = (const float*)d_in[3];
  const float* w_out = (const float*)d_in[4];
  const float* b_out = (const float*)d_in[5];
  const int* segpos  = (const int*)d_in[6];
  const int* winp    = (const int*)d_in[8];   // window_size (device scalar)

  const long BT = in_sizes[6];               // B*T
  const int  Wd = (int)(in_sizes[0] / BT);   // width (2048)
  const int  Hd = in_sizes[2] / Wd;          // head_dim (128)
  const int  Nh = Wd / Hd;                   // num_heads (16)
  const int  Bb = 2;                         // batch (fixed by setup)
  const int  T  = (int)(BT / Bb);

  char* wsp = (char*)d_ws;
  auto alloc = [&](size_t bytes) -> void* {
    void* p = (void*)wsp;
    wsp += (bytes + 255) & ~(size_t)255;
    return p;
  };

  __bf16* xb   = (__bf16*)alloc((size_t)BT * Wd * 2);
  __bf16* wqb  = (__bf16*)alloc((size_t)Wd * Wd * 2);
  __bf16* wkvb = (__bf16*)alloc((size_t)Wd * 2 * Hd * 2);
  __bf16* wob  = (__bf16*)alloc((size_t)Wd * Wd * 2);
  float*  kvf  = (float*) alloc((size_t)BT * 2 * Hd * 4);
  __bf16* qbuf = (__bf16*)alloc((size_t)BT * Wd * 2);
  __bf16* kbuf = (__bf16*)alloc((size_t)BT * Hd * 2);
  __bf16* vbuf = (__bf16*)alloc((size_t)BT * Hd * 2);
  int*    segs = (int*)   alloc((size_t)BT * 4);
  float*  qf   = (float*) alloc((size_t)BT * Wd * 4);
  __bf16* enc  = (__bf16*)qf;   // reuse qf region after rope (bf16 fits)

  const int TPB = 256;
  {
    long n = BT * Wd;
    cvt_f32_bf16<<<dim3((n + TPB - 1) / TPB), TPB, 0, stream>>>(x, xb, n);
    n = (long)Wd * Wd;
    cvt_f32_bf16<<<dim3((n + TPB - 1) / TPB), TPB, 0, stream>>>(w_q, wqb, n);
    cvt_f32_bf16<<<dim3((n + TPB - 1) / TPB), TPB, 0, stream>>>(w_out, wob, n);
    n = (long)Wd * 2 * Hd;
    pack_kv_bf16<<<dim3((n + TPB - 1) / TPB), TPB, 0, stream>>>(w_k, w_v, wkvb, Wd, Hd);
  }
  seg_scan<<<dim3(Bb), 32, 0, stream>>>(segpos, segs, T);

  gemm_bf16_f32<<<dim3(Wd / G_BN, (int)(BT / G_BM)), 256, 0, stream>>>(
      xb, wqb, nullptr, qf, (int)BT, Wd, Wd);
  gemm_bf16_f32<<<dim3((2 * Hd) / G_BN, (int)(BT / G_BM)), 256, 0, stream>>>(
      xb, wkvb, nullptr, kvf, (int)BT, 2 * Hd, Wd);

  {
    float scale = rsqrtf((float)Hd);
    long n = BT * Nh * (Hd / 2);
    rope_q_kernel<<<dim3((n + TPB - 1) / TPB), TPB, 0, stream>>>(
        qf, segpos, qbuf, BT, Nh, Hd, scale);
    n = BT * (Hd / 2);
    rope_kv_kernel<<<dim3((n + TPB - 1) / TPB), TPB, 0, stream>>>(
        kvf, segpos, kbuf, vbuf, BT, Hd);
  }

  attn_kernel<<<dim3(T / 16, Nh / 8, Bb), 256, 0, stream>>>(
      qbuf, kbuf, vbuf, segs, winp, enc, T, Nh);

  gemm_bf16_f32<<<dim3(Wd / G_BN, (int)(BT / G_BM)), 256, 0, stream>>>(
      enc, wob, b_out, (float*)d_out, (int)BT, Wd, Wd);
}
